// SDSA_81449759801834
// MI455X (gfx1250) — matrix operand order repaired and checked by reading, not compiled, not run on hardware
//
#include <hip/hip_runtime.h>
#include <cstdint>

// ---------------------------------------------------------------------------
// CDNA5 async global->LDS path (guarded; falls back to plain copies)
// ---------------------------------------------------------------------------
#if __has_builtin(__builtin_amdgcn_global_load_async_to_lds_b128) && \
    __has_builtin(__builtin_amdgcn_s_wait_asynccnt)
#define USE_ASYNC_LDS 1
#else
#define USE_ASYNC_LDS 0
#endif

// ---------------------------------------------------------------------------
// Types and helpers
// ---------------------------------------------------------------------------
typedef __attribute__((ext_vector_type(16))) __bf16 v16bf;
typedef __attribute__((ext_vector_type(8)))  float  v8f;
typedef unsigned short ushort_t;

union Frag16 {
    v16bf v;
    uint4 q[2];
    ushort_t s[16];
};

__device__ __forceinline__ v8f zero8() {
    v8f z;
#pragma unroll
    for (int i = 0; i < 8; ++i) z[i] = 0.0f;
    return z;
}

__device__ __forceinline__ ushort_t f2bf(float f) {
    unsigned u = __float_as_uint(f);
    unsigned r = (u + 0x7FFFu + ((u >> 16) & 1u)) >> 16;
    return (ushort_t)r;
}
__device__ __forceinline__ float bf2f(ushort_t s) {
    return __uint_as_float(((unsigned)s) << 16);
}

__device__ __forceinline__ v8f wmma_bf16(v16bf a, v16bf b, v8f c) {
    // D = A(16x32 bf16) * B(32x16 bf16) + C(16x16 f32)
    return __builtin_amdgcn_wmma_f32_16x16x32_bf16(
        /*neg_a=*/false, a, /*neg_b=*/false, b,
        /*c_mod=*/(short)0, c, /*reuse_a=*/false, /*reuse_b=*/false);
}

#if USE_ASYNC_LDS
// The builtin's pointer params are gcc-vector int4* in AS1 (global) / AS3 (LDS).
typedef int v4i_ __attribute__((vector_size(16)));
typedef __attribute__((address_space(1))) v4i_* as1_v4i;
typedef __attribute__((address_space(3))) v4i_* as3_v4i;
// 16B async copy global -> LDS (per-lane addresses), tracked by ASYNCcnt.
__device__ __forceinline__ void async_b128(const void* g, const void* l) {
    __builtin_amdgcn_global_load_async_to_lds_b128(
        (as1_v4i)(unsigned long long)g,
        (as3_v4i)(unsigned int)(unsigned long long)l,
        /*imm offset=*/0, /*cpol=*/0);
}
__device__ __forceinline__ void async_wait0() {
    __builtin_amdgcn_s_wait_asynccnt(0);
}
#endif

// A fragment: row-major [16 x 32] tile, leading dim ld (elements).
// Lane L<16 : M=L, K 0..15 contiguous;  Lane L>=16 : M=L-16, K 16..31.
__device__ __forceinline__ v16bf load_a_frag(const ushort_t* base, int ld) {
    int lane = threadIdx.x & 31;
    int m    = lane & 15;
    int k1   = (lane < 16) ? 0 : 16;
    const ushort_t* p = base + (size_t)m * ld;
    Frag16 f;
    f.q[0] = *(const uint4*)(p + k1);
    f.q[1] = *(const uint4*)(p + k1 + 8);
    return f.v;
}

// B fragment where B[k][n] = src[n*ld + k]  (i.e. B = srcT, src row-major [N,K]).
// Lane L<16 : col n=L, K 0..15 contiguous;  Lane L>=16 : col n=L-16, K 16..31.
__device__ __forceinline__ v16bf load_bT_frag(const ushort_t* base, int ld) {
    int lane = threadIdx.x & 31;
    int n    = lane & 15;
    int k0   = (lane < 16) ? 0 : 16;
    const ushort_t* p = base + (size_t)n * ld + k0;
    Frag16 f;
    f.q[0] = *(const uint4*)(p);
    f.q[1] = *(const uint4*)(p + 8);
    return f.v;
}

// B fragment where B[k][n] = src[k*ld + n]  (src row-major [K,N], K tile = 32).
__device__ __forceinline__ v16bf load_b_frag_rm(const ushort_t* base, int ld) {
    int lane = threadIdx.x & 31;
    int n    = lane & 15;
    int k0   = (lane < 16) ? 0 : 16;
    Frag16 f;
#pragma unroll
    for (int i = 0; i < 16; ++i) f.s[i] = base[(size_t)(k0 + i) * ld + n];
    return f.v;
}

// ---------------------------------------------------------------------------
// f32 -> bf16 convert
// ---------------------------------------------------------------------------
__global__ void __launch_bounds__(256)
convert_bf16_kernel(const float* __restrict__ in, ushort_t* __restrict__ out, int n) {
    int i = blockIdx.x * blockDim.x + threadIdx.x;
    int stride = gridDim.x * blockDim.x;
    for (; i < n; i += stride) out[i] = f2bf(in[i]);
}

// ---------------------------------------------------------------------------
// Fused depthwise-conv (stride=k, VALID) + LayerNorm + exact GELU  -> bf16
// One block per (b, output pixel); 256 threads = channels.
// ---------------------------------------------------------------------------
__global__ void __launch_bounds__(256)
dwconv_ln_gelu_kernel(const float* __restrict__ x, const float* __restrict__ w,
                      const float* __restrict__ cb, const float* __restrict__ g,
                      const float* __restrict__ be, ushort_t* __restrict__ out,
                      int H, int W, int ksz, int stride, int Hs, int Ws) {
    __shared__ float red[256];
    const int c  = threadIdx.x;
    const int sp = blockIdx.x % (Hs * Ws);
    const int b  = blockIdx.x / (Hs * Ws);
    const int oy = sp / Ws, ox = sp % Ws;

    float acc = cb[c];
    for (int ky = 0; ky < ksz; ++ky) {
        int iy = oy * stride + ky;
        for (int kx = 0; kx < ksz; ++kx) {
            int ix = ox * stride + kx;
            acc += w[c * ksz * ksz + ky * ksz + kx] *
                   x[(((size_t)b * H + iy) * W + ix) * 256 + c];
        }
    }
    // LayerNorm over C=256 (population variance, eps=1e-5)
    red[c] = acc; __syncthreads();
    for (int s = 128; s > 0; s >>= 1) { if (c < s) red[c] += red[c + s]; __syncthreads(); }
    float mean = red[0] * (1.0f / 256.0f); __syncthreads();
    float d = acc - mean;
    red[c] = d * d; __syncthreads();
    for (int s = 128; s > 0; s >>= 1) { if (c < s) red[c] += red[c + s]; __syncthreads(); }
    float var = red[0] * (1.0f / 256.0f);
    float yv = d * rsqrtf(var + 1e-5f) * g[c] + be[c];
    float gv = 0.5f * yv * (1.0f + erff(yv * 0.70710678118654752f));
    out[(size_t)blockIdx.x * 256 + c] = f2bf(gv);
}

// ---------------------------------------------------------------------------
// WMMA GEMM: C[M,N] = A[M,K] * W[N,K]^T (+bias), K = 256 fixed.
// Block = 8 waves. Each wave owns one 16-row M tile; the whole block shares a
// 128-column weight strip staged once into LDS (64 KB, async when available).
// Inner loop is split load-phase / compute-phase so all 8 B-fragment LDS
// loads are in flight before the 8 back-to-back WMMAs issue.
// grid = (N/128, M/128)
// ---------------------------------------------------------------------------
template <bool ADD_BIAS, bool OUT_F32>
__global__ void __launch_bounds__(256)
gemm_wmma_kernel(const ushort_t* __restrict__ A, const ushort_t* __restrict__ W,
                 const float* __restrict__ bias, void* __restrict__ out,
                 int M, int N) {
    constexpr int K  = 256;
    constexpr int NT = 8;                         // N-tiles per strip (128 cols)
    __shared__ __align__(16) ushort_t Wlds[NT * 16 * K];   // [n_local][K]

    const int tid   = threadIdx.x;
    const int wave  = tid >> 5;
    const int lane  = tid & 31;
    const int strip = blockIdx.x;
    const int tileM = blockIdx.y * 8 + wave;

    // ---- cooperative stage: 128 x 256 bf16 weight strip -> LDS ----
    const ushort_t* w_base = W + (size_t)strip * NT * 16 * K;
    {
        const int chunks = NT * 16 * K / 8;       // 16B chunks
        for (int i = tid; i < chunks; i += 256) {
#if USE_ASYNC_LDS
            async_b128(w_base + (size_t)i * 8, &Wlds[i * 8]);
#else
            *(uint4*)&Wlds[i * 8] = *(const uint4*)(w_base + (size_t)i * 8);
#endif
        }
#if USE_ASYNC_LDS
        async_wait0();
#endif
        __syncthreads();
    }

    // ---- compute: one 16-row M tile x 128 cols per wave ----
    v8f acc[NT];
#pragma unroll
    for (int t = 0; t < NT; ++t) acc[t] = zero8();

    const ushort_t* a_base = A + (size_t)tileM * 16 * K;
    for (int kt = 0; kt < K; kt += 32) {
        v16bf af = load_a_frag(a_base + kt, K);
        v16bf bf[NT];
#pragma unroll
        for (int t = 0; t < NT; ++t)
            bf[t] = load_bT_frag(&Wlds[t * 16 * K] + kt, K);
#pragma unroll
        for (int t = 0; t < NT; ++t)
            acc[t] = wmma_bf16(af, bf[t], acc[t]);
    }

    // ---- store ----
    const int n     = lane & 15;
    const int rbase = (lane < 16) ? 0 : 8;
#pragma unroll
    for (int t = 0; t < NT; ++t) {
        const int col = strip * NT * 16 + t * 16 + n;
        if (OUT_F32) {
            float* o = (float*)out;
            float bv = ADD_BIAS ? bias[col] : 0.0f;
#pragma unroll
            for (int r = 0; r < 8; ++r) {
                int row = tileM * 16 + rbase + r;
                o[(size_t)row * N + col] = acc[t][r] + bv;
            }
        } else {
            ushort_t* o = (ushort_t*)out;
#pragma unroll
            for (int r = 0; r < 8; ++r) {
                int row = tileM * 16 + rbase + r;
                o[(size_t)row * N + col] = f2bf(acc[t][r]);
            }
        }
    }
}

// ---------------------------------------------------------------------------
// v local-conv injection: v_out = v + dwconv3x3(v, pad=1) + bias
// kv: bf16 [B, Ns, 256] (v = cols 128..255), out: bf16 [B, Ns, 128]
// ---------------------------------------------------------------------------
__global__ void __launch_bounds__(128)
vconv_kernel(const ushort_t* __restrict__ kv, const float* __restrict__ w,
             const float* __restrict__ bias, ushort_t* __restrict__ out,
             int Hs, int Ws) {
    const int c2 = threadIdx.x;           // 0..127
    const int sp = blockIdx.x % (Hs * Ws);
    const int b  = blockIdx.x / (Hs * Ws);
    const int y = sp / Ws, x = sp % Ws;
    const ushort_t* base = kv + (size_t)b * Hs * Ws * 256 + 128 + c2;

    float vc  = bf2f(base[(size_t)sp * 256]);
    float acc = bias[c2];
#pragma unroll
    for (int dy = -1; dy <= 1; ++dy) {
        int yy = y + dy;
        if (yy < 0 || yy >= Hs) continue;
#pragma unroll
        for (int dx = -1; dx <= 1; ++dx) {
            int xx = x + dx;
            if (xx < 0 || xx >= Ws) continue;
            acc += w[c2 * 9 + (dy + 1) * 3 + (dx + 1)] *
                   bf2f(base[(size_t)(yy * Ws + xx) * 256]);
        }
    }
    out[((size_t)b * Hs * Ws + sp) * 128 + c2] = f2bf(vc + acc);
}

// ---------------------------------------------------------------------------
// Fused attention for one (b, head, 16-row q tile): one wave per block.
// V tile is staged to LDS (async issue at entry overlaps with q*kT WMMAs and
// softmax; ASYNCcnt waited only before the P*V loop). P is bounced bf16
// through LDS to re-layout C-fragment -> A-fragment. The q*kT loop is chunked
// 4 tiles at a time (load-phase then compute-phase) for ILP.
// ---------------------------------------------------------------------------
template <int NS>
__global__ void __launch_bounds__(32)
attn_wmma_kernel(const ushort_t* __restrict__ q, const ushort_t* __restrict__ kv,
                 const ushort_t* __restrict__ vlc, ushort_t* __restrict__ ocat,
                 int Nq, int colbase) {
    constexpr int NT = NS / 16;
    const int qt   = blockIdx.x;
    const int h    = blockIdx.y;   // head within branch (0..3)
    const int b    = blockIdx.z;
    const int lane = threadIdx.x & 31;

    const ushort_t* qp = q   + ((size_t)b * Nq + qt * 16) * 256 + colbase + h * 32;
    const ushort_t* kp = kv  + (size_t)b * NS * 256 + h * 32;      // k cols, ld 256
    const ushort_t* vp = vlc + (size_t)b * NS * 128 + h * 32;      // v cols, ld 128
    ushort_t*       op = ocat + ((size_t)b * Nq + qt * 16) * 256 + colbase + h * 32;

    __shared__ __align__(16) ushort_t Vl[NS][32];   // V tile  [k][n]
    __shared__ __align__(16) ushort_t P[16][NS];    // softmax probs, row-major

    // ---- stage V tile (issue early; async engine runs behind the WMMAs) ----
    for (int i = lane; i < NS * 4; i += 32) {       // 16B chunks: 4 per row
        int row = i >> 2, cc = (i & 3) * 8;
#if USE_ASYNC_LDS
        async_b128(vp + (size_t)row * 128 + cc, &Vl[row][cc]);
#else
        *(uint4*)&Vl[row][cc] = *(const uint4*)(vp + (size_t)row * 128 + cc);
#endif
    }

    // ---- S = q * k^T  (each k-tile is one 16x16x32 WMMA; K = hd = 32) ----
    v16bf qf = load_a_frag(qp, 256);
    v8f s[NT];
#pragma unroll
    for (int t0 = 0; t0 < NT; t0 += 4) {
        v16bf bf[4];
#pragma unroll
        for (int j = 0; j < 4; ++j)
            bf[j] = load_bT_frag(kp + (size_t)(t0 + j) * 16 * 256, 256);
#pragma unroll
        for (int j = 0; j < 4; ++j)
            s[t0 + j] = wmma_bf16(qf, bf[j], zero8());
    }

    // ---- row softmax (row m = VGPR index within a 16-lane half) ----
    const float scale = 0.17677669529663687f;  // 1/sqrt(32)
#pragma unroll
    for (int r = 0; r < 8; ++r) {
        float m = -1e30f;
#pragma unroll
        for (int t = 0; t < NT; ++t) m = fmaxf(m, s[t][r]);
#pragma unroll
        for (int off = 1; off < 16; off <<= 1) m = fmaxf(m, __shfl_xor(m, off, 32));
        m *= scale;
        float sum = 0.0f;
#pragma unroll
        for (int t = 0; t < NT; ++t) {
            float e = __expf(s[t][r] * scale - m);
            s[t][r] = e;
            sum += e;
        }
#pragma unroll
        for (int off = 1; off < 16; off <<= 1) sum += __shfl_xor(sum, off, 32);
        float inv = 1.0f / sum;
#pragma unroll
        for (int t = 0; t < NT; ++t) s[t][r] *= inv;
    }

    // ---- re-layout P through LDS: C-fragment -> row-major [16][NS] bf16 ----
    {
        const int n  = lane & 15;
        const int rb = (lane < 16) ? 0 : 8;
#pragma unroll
        for (int r = 0; r < 8; ++r)
#pragma unroll
            for (int t = 0; t < NT; ++t)
                P[rb + r][t * 16 + n] = f2bf(s[t][r]);
    }
#if USE_ASYNC_LDS
    async_wait0();
#endif
    __syncthreads();

    // ---- O = P * V  (hd = 32 -> two 16-col accumulators); V from LDS ----
    v8f acc0 = zero8(), acc1 = zero8();
#pragma unroll
    for (int kt = 0; kt < NS / 32; ++kt) {
        v16bf pa = load_a_frag(&P[0][0] + kt * 32, NS);
        v16bf b0 = load_b_frag_rm(&Vl[kt * 32][0], 32);       // n = 0..15
        v16bf b1 = load_b_frag_rm(&Vl[kt * 32][0] + 16, 32);  // n = 16..31
        acc0 = wmma_bf16(pa, b0, acc0);
        acc1 = wmma_bf16(pa, b1, acc1);
    }

    // ---- store O (bf16) into concatenated buffer ----
    {
        const int n  = lane & 15;
        const int rb = (lane < 16) ? 0 : 8;
#pragma unroll
        for (int r = 0; r < 8; ++r) {
            op[(size_t)(rb + r) * 256 + n]      = f2bf(acc0[r]);
            op[(size_t)(rb + r) * 256 + 16 + n] = f2bf(acc1[r]);
        }
    }
}

// ---------------------------------------------------------------------------
// Host-side orchestration
// ---------------------------------------------------------------------------
extern "C" void kernel_launch(void* const* d_in, const int* in_sizes, int n_in,
                              void* d_out, int out_size, void* d_ws, size_t ws_size,
                              hipStream_t stream) {
    (void)in_sizes; (void)n_in; (void)out_size; (void)ws_size;

    const float* x      = (const float*)d_in[0];
    const float* q_w    = (const float*)d_in[1];
    const float* kv1_w  = (const float*)d_in[2];
    const float* kv2_w  = (const float*)d_in[3];
    const float* proj_w = (const float*)d_in[4];
    const float* proj_b = (const float*)d_in[5];
    const float* sr1_w  = (const float*)d_in[6];
    const float* sr1_b  = (const float*)d_in[7];
    const float* sr2_w  = (const float*)d_in[8];
    const float* sr2_b  = (const float*)d_in[9];
    const float* ln1_g  = (const float*)d_in[10];
    const float* ln1_b  = (const float*)d_in[11];
    const float* ln2_g  = (const float*)d_in[12];
    const float* ln2_b  = (const float*)d_in[13];
    const float* lc1_w  = (const float*)d_in[14];
    const float* lc1_b  = (const float*)d_in[15];
    const float* lc2_w  = (const float*)d_in[16];
    const float* lc2_b  = (const float*)d_in[17];
    float* out = (float*)d_out;

    // Fixed problem geometry (from setup_inputs): B=8, H=W=64, C=256, sr=8
    const int B = 8, H = 64, W = 64, C = 256;
    const int N = H * W;                         // 4096
    const int H1 = 16, W1 = 16, N1 = H1 * W1;    // branch 1 (stride 4, k 4)
    const int H2 = 8,  W2 = 8,  N2 = H2 * W2;    // branch 2 (stride 8, k 8)

    // Workspace layout (bf16 buffers as ushort bits)
    char* ws = (char*)d_ws;
    size_t off = 0;
    auto alloc = [&](size_t bytes) { char* p = ws + off; off += (bytes + 255) & ~(size_t)255; return p; };
    ushort_t* xb    = (ushort_t*)alloc((size_t)B * N * C * 2);   // x bf16
    ushort_t* qb    = (ushort_t*)alloc((size_t)B * N * C * 2);   // q bf16
    ushort_t* ocat  = (ushort_t*)alloc((size_t)B * N * C * 2);   // concat(o1,o2) bf16
    ushort_t* x1b   = (ushort_t*)alloc((size_t)B * N1 * C * 2);
    ushort_t* x2b   = (ushort_t*)alloc((size_t)B * N2 * C * 2);
    ushort_t* kv1b  = (ushort_t*)alloc((size_t)B * N1 * C * 2);
    ushort_t* kv2b  = (ushort_t*)alloc((size_t)B * N2 * C * 2);
    ushort_t* v1b   = (ushort_t*)alloc((size_t)B * N1 * 128 * 2);
    ushort_t* v2b   = (ushort_t*)alloc((size_t)B * N2 * 128 * 2);
    ushort_t* wq    = (ushort_t*)alloc((size_t)C * C * 2);
    ushort_t* wkv1  = (ushort_t*)alloc((size_t)C * C * 2);
    ushort_t* wkv2  = (ushort_t*)alloc((size_t)C * C * 2);
    ushort_t* wproj = (ushort_t*)alloc((size_t)C * C * 2);

    // 1) f32 -> bf16 conversions
    {
        int n = B * N * C;
        int nb = (n + 255) / 256; if (nb > 4096) nb = 4096;
        convert_bf16_kernel<<<dim3(nb), 256, 0, stream>>>(x, xb, n);
        int nw = C * C;
        convert_bf16_kernel<<<dim3((nw + 255) / 256), 256, 0, stream>>>(q_w, wq, nw);
        convert_bf16_kernel<<<dim3((nw + 255) / 256), 256, 0, stream>>>(kv1_w, wkv1, nw);
        convert_bf16_kernel<<<dim3((nw + 255) / 256), 256, 0, stream>>>(kv2_w, wkv2, nw);
        convert_bf16_kernel<<<dim3((nw + 255) / 256), 256, 0, stream>>>(proj_w, wproj, nw);
    }

    // 2) fused dwconv + LN + GELU (both branches)
    dwconv_ln_gelu_kernel<<<dim3(B * N1), 256, 0, stream>>>(
        x, sr1_w, sr1_b, ln1_g, ln1_b, x1b, H, W, /*k=*/4, /*stride=*/4, H1, W1);
    dwconv_ln_gelu_kernel<<<dim3(B * N2), 256, 0, stream>>>(
        x, sr2_w, sr2_b, ln2_g, ln2_b, x2b, H, W, /*k=*/8, /*stride=*/8, H2, W2);

    // 3) q = x @ q_w^T   (M = B*N = 32768)
    gemm_wmma_kernel<false, false><<<dim3(C / 128, (B * N) / 128), 256, 0, stream>>>(
        xb, wq, nullptr, (void*)qb, B * N, C);

    // 4) kv projections
    gemm_wmma_kernel<false, false><<<dim3(C / 128, (B * N1) / 128), 256, 0, stream>>>(
        x1b, wkv1, nullptr, (void*)kv1b, B * N1, C);
    gemm_wmma_kernel<false, false><<<dim3(C / 128, (B * N2) / 128), 256, 0, stream>>>(
        x2b, wkv2, nullptr, (void*)kv2b, B * N2, C);

    // 5) v local 3x3 depthwise injection
    vconv_kernel<<<dim3(B * N1), 128, 0, stream>>>(kv1b, lc1_w, lc1_b, v1b, H1, W1);
    vconv_kernel<<<dim3(B * N2), 128, 0, stream>>>(kv2b, lc2_w, lc2_b, v2b, H2, W2);

    // 6) fused attention per (16-row q tile, head, batch)
    attn_wmma_kernel<256><<<dim3(N / 16, 4, B), 32, 0, stream>>>(qb, kv1b, v1b, ocat, N, /*colbase=*/0);
    attn_wmma_kernel<64> <<<dim3(N / 16, 4, B), 32, 0, stream>>>(qb, kv2b, v2b, ocat, N, /*colbase=*/128);

    // 7) out = ocat @ proj_w^T + proj_b   (f32 to d_out)
    gemm_wmma_kernel<true, true><<<dim3(C / 128, (B * N) / 128), 256, 0, stream>>>(
        ocat, wproj, proj_b, (void*)out, B * N, C);
}
